// RN_79542794322147
// MI455X (gfx1250) — matrix operand order repaired and checked by reading, not compiled, not run on hardware
//
#include <hip/hip_runtime.h>
#include <cstdint>
#include <cstddef>

typedef __attribute__((ext_vector_type(16))) __bf16 v16bf;
typedef __attribute__((ext_vector_type(8)))  float  v8f;

#define NR 100
#define BS 32

__device__ __forceinline__ unsigned short f2bf(float f) {
  unsigned int u = __float_as_uint(f);
  unsigned int r = 0x7FFFu + ((u >> 16) & 1u);
  return (unsigned short)((u + r) >> 16);
}
__device__ __forceinline__ float bf2f(unsigned short h) {
  return __uint_as_float(((unsigned int)h) << 16);
}

// ---------------- Projection: H = relu(Xp @ Wv + bv), 4 rows per block ----------------
__global__ __launch_bounds__(256)
void projx_k(const float* __restrict__ X, const float* __restrict__ pos,
             const float* __restrict__ Wv, const float* __restrict__ bv,
             float* __restrict__ H) {
  int rg = blockIdx.x;           // row group: rows rg*4 .. rg*4+3  (0..799)
  int c = threadIdx.x;           // 0..255
  const float* x0 = X + (size_t)rg * 4 * 2048;
  float acc[4];
#pragma unroll
  for (int r = 0; r < 4; ++r) acc[r] = bv[c];
  for (int k = 0; k < 2048; ++k) {
    float w = Wv[(size_t)k * 256 + c];
#pragma unroll
    for (int r = 0; r < 4; ++r) acc[r] = fmaf(x0[(size_t)r * 2048 + k], w, acc[r]);
  }
#pragma unroll
  for (int r = 0; r < 4; ++r) {
    const float* p = pos + (size_t)(rg * 4 + r) * 6;  // pe = pos[:,:, [0,1,4,5]]
    acc[r] = fmaf(p[0], Wv[(size_t)2048 * 256 + c], acc[r]);
    acc[r] = fmaf(p[1], Wv[(size_t)2049 * 256 + c], acc[r]);
    acc[r] = fmaf(p[4], Wv[(size_t)2050 * 256 + c], acc[r]);
    acc[r] = fmaf(p[5], Wv[(size_t)2051 * 256 + c], acc[r]);
    H[(size_t)(rg * 4 + r) * 256 + c] = fmaxf(acc[r], 0.f);
  }
}

__global__ __launch_bounds__(256)
void projq_k(const float* __restrict__ Q, const float* __restrict__ Wq,
             const float* __restrict__ bq, float* __restrict__ Qv) {
  int b = blockIdx.x;            // 0..31
  int c = threadIdx.x;
  const float* q = Q + (size_t)b * 1024;
  float acc = bq[c];
  for (int k = 0; k < 1024; ++k) acc = fmaf(q[k], Wq[(size_t)k * 256 + c], acc);
  Qv[(size_t)b * 256 + c] = fmaxf(acc, 0.f);
}

// ------------- pair[b,i,j,c] = (H[b,i,c]+Qv)*(H[b,j,c]+Qv), bf16 NHWC -------------
__global__ __launch_bounds__(256)
void pair_k(const float* __restrict__ H, const float* __restrict__ Qv,
            unsigned short* __restrict__ pr) {
  size_t pix = blockIdx.x;       // b*10000 + i*100 + j
  int c = threadIdx.x;
  int b = (int)(pix / 10000);
  int ij = (int)(pix % 10000);
  int i = ij / 100, j = ij % 100;
  float q = Qv[b * 256 + c];
  float hi = H[((size_t)(b * 100 + i)) * 256 + c] + q;
  float hj = H[((size_t)(b * 100 + j)) * 256 + c] + q;
  pr[pix * 256 + c] = f2bf(hi * hj);
}

// ------------- Pack OIHW f32 weights into WMMA A-register order (bf16) -------------
// block = ((coT*3+kh)*3+kw)*CT + ct ; block holds 32 lanes x 8 dwords.
// A order used here (contiguous k, same as B): lane<16 -> M=lane, K = 2*r+h
//                          lane>=16 -> M=lane-16, K = 16+2*r+h
__global__ __launch_bounds__(256)
void pack_w(const float* __restrict__ W, unsigned int* __restrict__ apack,
            int COUT, int CIN) {
  int total = (COUT / 16) * 9 * (CIN / 32) * 256;
  int idx = blockIdx.x * 256 + threadIdx.x;
  if (idx >= total) return;
  int dw = idx & 255;
  int blk = idx >> 8;
  int lane = dw >> 3;
  int r = dw & 7;
  int CT = CIN / 32;
  int ct = blk % CT;
  int rest = blk / CT;
  int kw = rest % 3; rest /= 3;
  int kh = rest % 3;
  int coT = rest / 3;
  int m = lane & 15;
  int kb = ((lane >> 4) ? 16 : 0) + 2 * r;
  int co = coT * 16 + m;
  int c0 = ct * 32 + kb;
  float w0 = W[((size_t)(co * CIN + c0) * 3 + kh) * 3 + kw];
  float w1 = W[((size_t)(co * CIN + c0 + 1) * 3 + kh) * 3 + kw];
  apack[idx] = (unsigned int)f2bf(w0) | ((unsigned int)f2bf(w1) << 16);
}

// ------------- Implicit-GEMM dilated 3x3 conv via v_wmma_f32_16x16x32_bf16 -------------
// M-blocked: each wave accumulates MBLK Cout-tiles (M = MBLK*16) for one 16-pixel tile,
// so every B (activation) tile feeds MBLK WMMAs.
// in:  [BS, NR, NR, CIN]  bf16 channel-last
// out: [BS, NR, NR, COUT] bf16 channel-last, relu(conv + bias)
template<int CIN, int COUT, int DIL, int MBLK>
__global__ __launch_bounds__(32)
void conv_wmma(const unsigned short* __restrict__ in,
               const unsigned int* __restrict__ apack,
               const float* __restrict__ bias,
               unsigned short* __restrict__ out) {
  constexpr int JT = 7;                 // ceil(100/16) pixel tiles per row
  constexpr int CT = CIN / 32;
  const int tile = blockIdx.x;          // 0..NR*JT-1
  const int coB  = blockIdx.y * MBLK;   // base Cout tile
  const int b    = blockIdx.z;
  const int i    = tile / JT;
  const int j0   = (tile - i * JT) * 16;
  const int lane = threadIdx.x;
  const int n    = lane & 15;           // B/D column = output pixel within tile
  const int hi   = lane >> 4;

  const v8f zero = {};
  v8f acc[MBLK];
#pragma unroll
  for (int m = 0; m < MBLK; ++m) acc[m] = zero;

  for (int kh = 0; kh < 3; ++kh) {
    const int y = i + DIL * (kh - 1);
    const bool yok = (unsigned)y < (unsigned)NR;
    for (int kw = 0; kw < 3; ++kw) {
      const int x = j0 + n + DIL * (kw - 1);
      const bool ok = yok && ((unsigned)x < (unsigned)NR);
      // B tile base: 16 contiguous channels per lane (K=0..15 / 16..31)
      const unsigned short* bp = in
          + ((size_t)((b * NR + (yok ? y : 0)) * NR + (ok ? x : 0))) * CIN
          + hi * 16;
#pragma unroll
      for (int ct = 0; ct < CT; ++ct) {
        v16bf Bv = {};
        if (ok) Bv = *(const v16bf*)(bp + ct * 32);
#pragma unroll
        for (int m = 0; m < MBLK; ++m) {
          const unsigned int* ap = apack
              + ((size_t)(((((coB + m) * 3 + kh) * 3 + kw) * CT) + ct)) * 256
              + lane * 8;
          v16bf Av = *(const v16bf*)ap;
          acc[m] = __builtin_amdgcn_wmma_f32_16x16x32_bf16(
              false, Av, false, Bv, (short)0, acc[m], false, false);
        }
      }
    }
  }

  const int j = j0 + n;
  if (j < NR) {
    const size_t pixo = ((size_t)((b * NR + i) * NR + j)) * COUT;
#pragma unroll
    for (int m = 0; m < MBLK; ++m) {
      // C/D f32 layout: VGPR r -> channel co0 + r, 8 consecutive channels per lane
      const int co0 = (coB + m) * 16 + hi * 8;
      unsigned int w[4];
#pragma unroll
      for (int r = 0; r < 4; ++r) {
        float v0 = acc[m][2 * r]     + bias[co0 + 2 * r];
        float v1 = acc[m][2 * r + 1] + bias[co0 + 2 * r + 1];
        v0 = v0 > 0.f ? v0 : 0.f;
        v1 = v1 > 0.f ? v1 : 0.f;
        w[r] = (unsigned int)f2bf(v0) | ((unsigned int)f2bf(v1) << 16);
      }
      unsigned int* op = (unsigned int*)(out + pixo + co0);
      op[0] = w[0]; op[1] = w[1]; op[2] = w[2]; op[3] = w[3];
    }
  }
}

// ------------- Last conv layer: Cout=2, Cin=64, d=4, f32 logits [b,g,i,j] -------------
__global__ __launch_bounds__(256)
void conv3_k(const unsigned short* __restrict__ in, const float* __restrict__ W,
             const float* __restrict__ bias, float* __restrict__ out) {
  int idx = blockIdx.x * 256 + threadIdx.x;
  if (idx >= BS * NR * NR) return;
  int j = idx % NR; int t = idx / NR; int i = t % NR; int b = t / NR;
  float a0 = bias[0], a1 = bias[1];
  for (int kh = 0; kh < 3; ++kh) {
    int y = i + 4 * (kh - 1);
    if ((unsigned)y >= (unsigned)NR) continue;
    for (int kw = 0; kw < 3; ++kw) {
      int x = j + 4 * (kw - 1);
      if ((unsigned)x >= (unsigned)NR) continue;
      const unsigned short* p = in + ((size_t)((b * NR + y) * NR + x)) * 64;
      const float* w0 = W + kh * 3 + kw;             // W[0,c,kh,kw]
      const float* w1 = W + (size_t)64 * 9 + kh * 3 + kw;
      for (int c = 0; c < 64; ++c) {
        float v = bf2f(p[c]);
        a0 = fmaf(v, w0[c * 9], a0);
        a1 = fmaf(v, w1[c * 9], a1);
      }
    }
  }
  out[((size_t)(b * 2 + 0) * NR + i) * NR + j] = fmaxf(a0, 0.f);
  out[((size_t)(b * 2 + 1) * NR + i) * NR + j] = fmaxf(a1, 0.f);
}

// ------------- sym + softmax over flattened NrxNr per (b,g) -------------
__global__ __launch_bounds__(256)
void symsm_k(const float* __restrict__ logits, float* __restrict__ out) {
  int bg = blockIdx.x;           // b*2+g, 0..63
  const float* L = logits + (size_t)bg * (NR * NR);
  float* O = out + (size_t)bg * (NR * NR);
  int tid = threadIdx.x;
  __shared__ float red[256];

  float mx = -3.4e38f;
  for (int e = tid; e < NR * NR; e += 256) {
    int i = e / NR, j = e - i * NR;
    float s = L[e] + L[j * NR + i];
    mx = fmaxf(mx, s);
  }
  red[tid] = mx; __syncthreads();
  for (int s = 128; s > 0; s >>= 1) {
    if (tid < s) red[tid] = fmaxf(red[tid], red[tid + s]);
    __syncthreads();
  }
  mx = red[0]; __syncthreads();

  float sum = 0.f;
  for (int e = tid; e < NR * NR; e += 256) {
    int i = e / NR, j = e - i * NR;
    sum += __expf(L[e] + L[j * NR + i] - mx);
  }
  red[tid] = sum; __syncthreads();
  for (int s = 128; s > 0; s >>= 1) {
    if (tid < s) red[tid] += red[tid + s];
    __syncthreads();
  }
  float inv = 1.f / red[0];

  for (int e = tid; e < NR * NR; e += 256) {
    int i = e / NR, j = e - i * NR;
    O[e] = __expf(L[e] + L[j * NR + i] - mx) * inv;
  }
}

// ------------- S[b,i,j] = 0.25 * sum_g (rm0+rm1)[b,g,i,j] -------------
__global__ __launch_bounds__(256)
void sum_rm_k(const float* __restrict__ rm0, const float* __restrict__ rm1,
              float* __restrict__ S) {
  int idx = blockIdx.x * 256 + threadIdx.x;
  if (idx >= BS * NR * NR) return;
  int b = idx / (NR * NR);
  int e = idx - b * (NR * NR);
  size_t g0 = (size_t)(b * 2) * (NR * NR) + e;
  size_t g1 = g0 + NR * NR;
  S[idx] = 0.25f * (rm0[g0] + rm0[g1] + rm1[g0] + rm1[g1]);
}

// ------------- Y[b,i,d] = sum_j S[b,i,j] * X[b,j,d], 4 rows per block -------------
__global__ __launch_bounds__(256)
void agg_k(const float* __restrict__ S, const float* __restrict__ X,
           float* __restrict__ Y) {
  __shared__ float sS[4][NR];
  int big = blockIdx.x;                   // row group: rows big*4..+3 (0..799)
  int d = blockIdx.y * 256 + threadIdx.x; // 0..2047
  if (threadIdx.x < NR) {
#pragma unroll
    for (int r = 0; r < 4; ++r)
      sS[r][threadIdx.x] = S[(size_t)(big * 4 + r) * NR + threadIdx.x];
  }
  __syncthreads();
  int b = (big * 4) / NR;                 // 100 % 4 == 0 -> same b for all 4 rows
  const float* Xb = X + (size_t)b * NR * 2048;
  float acc[4] = {0.f, 0.f, 0.f, 0.f};
  for (int j = 0; j < NR; ++j) {
    float xv = Xb[(size_t)j * 2048 + d];
#pragma unroll
    for (int r = 0; r < 4; ++r) acc[r] = fmaf(sS[r][j], xv, acc[r]);
  }
#pragma unroll
  for (int r = 0; r < 4; ++r) Y[(size_t)(big * 4 + r) * 2048 + d] = acc[r];
}

// ----------------------------------------------------------------------
extern "C" void kernel_launch(void* const* d_in, const int* in_sizes, int n_in,
                              void* d_out, int out_size, void* d_ws, size_t ws_size,
                              hipStream_t stream) {
  // setup_inputs() dict insertion order:
  const float* X   = (const float*)d_in[0];
  const float* Q   = (const float*)d_in[1];
  const float* pos = (const float*)d_in[2];
  const float* Wv  = (const float*)d_in[3];
  const float* bv  = (const float*)d_in[4];
  const float* Wq  = (const float*)d_in[5];
  const float* bq  = (const float*)d_in[6];
  const float* W01 = (const float*)d_in[7];
  const float* b01 = (const float*)d_in[8];
  const float* W1  = (const float*)d_in[9];
  const float* b1  = (const float*)d_in[10];
  const float* W02 = (const float*)d_in[11];
  const float* b02 = (const float*)d_in[12];
  const float* W2  = (const float*)d_in[13];
  const float* b2  = (const float*)d_in[14];
  const float* W03 = (const float*)d_in[15];
  const float* b03 = (const float*)d_in[16];
  const float* W3  = (const float*)d_in[17];
  const float* b3  = (const float*)d_in[18];

  char* ws = (char*)d_ws;
  size_t off = 0;
  auto alloc = [&](size_t bytes) -> char* {
    char* p = ws + off;
    off = (off + bytes + 255) & ~((size_t)255);
    return p;
  };

  float*          Hx    = (float*)alloc((size_t)BS * NR * 256 * 4);           // 3.3 MB
  float*          Qv    = (float*)alloc((size_t)BS * 256 * 4);
  unsigned short* pairB = (unsigned short*)alloc((size_t)BS * NR * NR * 256 * 2); // 164 MB
  unsigned int*   ap01  = (unsigned int*)alloc((size_t)8 * 9 * 8 * 1024);     // 576 KB
  unsigned int*   ap02  = (unsigned int*)alloc((size_t)4 * 9 * 4 * 1024);     // 144 KB
  unsigned int*   ap1   = (unsigned int*)alloc((size_t)8 * 9 * 8 * 1024);
  unsigned int*   ap2   = (unsigned int*)alloc((size_t)4 * 9 * 4 * 1024);
  unsigned short* act1  = (unsigned short*)alloc((size_t)BS * NR * NR * 128 * 2); // 82 MB
  unsigned short* act2  = (unsigned short*)alloc((size_t)BS * NR * NR * 64 * 2);  // 41 MB
  float*          logit = (float*)alloc((size_t)BS * 2 * NR * NR * 4);        // 2.6 MB
  float*          rm0   = (float*)alloc((size_t)BS * 2 * NR * NR * 4);
  float*          Ssum  = (float*)alloc((size_t)BS * NR * NR * 4);

  float* rm_out = (float*)d_out;                              // [32,2,100,100]
  float* relX   = (float*)d_out + (size_t)BS * 2 * NR * NR;   // [32,100,2048]

  // 1) projections + H
  projx_k<<<dim3(BS * NR / 4), 256, 0, stream>>>(X, pos, Wv, bv, Hx);
  projq_k<<<dim3(BS), 256, 0, stream>>>(Q, Wq, bq, Qv);

  // 2) pairwise tensor (bf16 NHWC)
  pair_k<<<dim3(BS * NR * NR), 256, 0, stream>>>(Hx, Qv, pairB);

  // 3) weight packing for WMMA A-operand
  pack_w<<<dim3(576), 256, 0, stream>>>(W01, ap01, 128, 256);
  pack_w<<<dim3(144), 256, 0, stream>>>(W02, ap02, 64, 128);
  pack_w<<<dim3(576), 256, 0, stream>>>(W1,  ap1,  128, 256);
  pack_w<<<dim3(144), 256, 0, stream>>>(W2,  ap2,  64, 128);

  const dim3 cg1(NR * 7, (128 / 16) / 4, BS);   // conv1: M-blocked by 4
  const dim3 cg2(NR * 7, (64 / 16) / 4, BS);    // conv2: M-blocked by 4
  const int  p3 = (BS * NR * NR + 255) / 256;

  // 4) stack 0 -> rm0
  conv_wmma<256, 128, 1, 4><<<cg1, 32, 0, stream>>>(pairB, ap01, b01, act1);
  conv_wmma<128, 64, 2, 4><<<cg2, 32, 0, stream>>>(act1, ap02, b02, act2);
  conv3_k<<<dim3(p3), 256, 0, stream>>>(act2, W03, b03, logit);
  symsm_k<<<dim3(BS * 2), 256, 0, stream>>>(logit, rm0);

  // 5) stack 1 -> rm (written straight into d_out)
  conv_wmma<256, 128, 1, 4><<<cg1, 32, 0, stream>>>(pairB, ap1, b1, act1);
  conv_wmma<128, 64, 2, 4><<<cg2, 32, 0, stream>>>(act1, ap2, b2, act2);
  conv3_k<<<dim3(p3), 256, 0, stream>>>(act2, W3, b3, logit);
  symsm_k<<<dim3(BS * 2), 256, 0, stream>>>(logit, rm_out);

  // 6) aggregate: relational_X = (rm + rm0) @ X / (2G)
  sum_rm_k<<<dim3(p3), 256, 0, stream>>>(rm0, rm_out, Ssum);
  agg_k<<<dim3(BS * NR / 4, 2048 / 256), 256, 0, stream>>>(Ssum, X, relX);
}
